// DMRInduction_12713103197275
// MI455X (gfx1250) — matrix operand order repaired and checked by reading, not compiled, not run on hardware
//
#include <hip/hip_runtime.h>
#include <hip/hip_bf16.h>
#include <cmath>

// ---------------- problem constants (match reference setup_inputs) ----------
#define IN_CAPS   128          // I
#define NUM_CAPS  5            // C
#define DIM_CAPS  153          // D
#define DPAD      160          // D padded to /32 for WMMA K
#define IN_DIM    768          // K of projection
#define NQ        256          // Q
#define NROWS     (IN_CAPS + NQ)        // 384 stacked [m;q] rows
#define NOUT      (NUM_CAPS*DIM_CAPS)   // 765
#define NPAD      768          // 765 padded to /16 tiles
#define EPSF      1e-8f

typedef __attribute__((ext_vector_type(16))) _Float16 v16h;
typedef __attribute__((ext_vector_type(8)))  _Float16 v8h;
typedef __attribute__((ext_vector_type(8)))  float    v8f;

// ---------------- WMMA helpers ---------------------------------------------
__device__ __forceinline__ v8f wmma_f16(v16h a, v16h b, v8f c) {
  // D = A(16x32 f16) x B(32x16 f16) + C(16x16 f32)
  return __builtin_amdgcn_wmma_f32_16x16x32_f16(
      /*neg_a=*/false, a, /*neg_b=*/false, b,
      /*c_mod=*/(short)0, c, /*reuse_a=*/false, /*reuse_b=*/false);
}

// A-matrix fragment (16x32, MxK) from row-major f16 [rows][ld], K contiguous.
// Per ISA: lanes 0-15 row M=lane, elems 0..7 = K(k0+h*8..), 8..15 = K(+16..)
__device__ __forceinline__ v16h fragA_f16(const _Float16* __restrict__ base,
                                          int ld, int row0, int k0) {
  const int lane = threadIdx.x & 31;
  const int r = lane & 15, h = lane >> 4;
  const _Float16* p = base + (size_t)(row0 + r) * ld + k0 + h * 8;
  v8h lo = *(const v8h*)(p);
  v8h hi = *(const v8h*)(p + 16);
  v16h f;
#pragma unroll
  for (int j = 0; j < 8; ++j) { f[j] = lo[j]; f[8 + j] = hi[j]; }
  return f;
}

// B-matrix fragment (32x16, KxN) where memory holds B^T row-major: [n][k],
// i.e. element B[k][n] = base[n*ld + k]. Same K order as fragA: lane col
// n = lane%16, elems 0..7 = K(k0+h*8..), 8..15 = K(+16..).
__device__ __forceinline__ v16h fragB_f16(const _Float16* __restrict__ base,
                                          int ld, int n0, int k0) {
  const int lane = threadIdx.x & 31;
  const int n = lane & 15, h = lane >> 4;
  const _Float16* p = base + (size_t)(n0 + n) * ld + k0 + h * 8;
  v8h lo = *(const v8h*)(p);
  v8h hi = *(const v8h*)(p + 16);
  v16h f;
#pragma unroll
  for (int j = 0; j < 8; ++j) { f[j] = lo[j]; f[8 + j] = hi[j]; }
  return f;
}

__device__ __forceinline__ float wave_sum(float s) {
#pragma unroll
  for (int off = 16; off; off >>= 1) s += __shfl_xor(s, off, 32);
  return s;
}

// ---------------- staging kernels -------------------------------------------

// AH[row][k] f16 = [m;q] rows, coalesced convert.
__global__ void k_prepA(const float* __restrict__ m, const float* __restrict__ q,
                        _Float16* __restrict__ AH) {
  int idx = blockIdx.x * blockDim.x + threadIdx.x;
  if (idx >= NROWS * IN_DIM) return;
  const int k = idx % IN_DIM, row = idx / IN_DIM;
  const float v = (row < IN_CAPS) ? m[(size_t)row * IN_DIM + k]
                                  : q[(size_t)(row - IN_CAPS) * IN_DIM + k];
  AH[idx] = (_Float16)v;
}

// WH[n][k] f16 = W[k][n] (transposed via LDS tile), cols 765..767 zero.
__global__ void k_prepW(const float* __restrict__ W, _Float16* __restrict__ WH) {
  __shared__ _Float16 tile[16][17];            // +1 to dodge bank conflicts
  const int k0 = blockIdx.x * 16, n0 = blockIdx.y * 16;
  const int tx = threadIdx.x & 15, ty = threadIdx.x >> 4;   // 256 threads
  {
    const int k = k0 + ty, n = n0 + tx;        // coalesced read along n
    const float v = (n < NOUT) ? W[(size_t)k * NOUT + n] : 0.f;
    tile[ty][tx] = (_Float16)v;
  }
  __syncthreads();
  // coalesced write along k
  WH[(size_t)(n0 + ty) * IN_DIM + k0 + tx] = tile[tx][ty];
}

// ---------------- compute kernels -------------------------------------------

// hat[0:128] = m@W+b ; hat[128:384] = q@W+b ; cols padded 765->768 (zero)
__global__ void k_proj(const _Float16* __restrict__ AH, const _Float16* __restrict__ WH,
                       const float* __restrict__ bias, float* __restrict__ hat) {
  const int nt = blockIdx.x;          // 0..47
  const int mt = blockIdx.y;          // 0..23
  const int row0 = mt * 16, col0 = nt * 16;
  v8f acc = {};
#pragma unroll 4
  for (int kc = 0; kc < IN_DIM; kc += 32)
    acc = wmma_f16(fragA_f16(AH, IN_DIM, row0, kc),
                   fragB_f16(WH, IN_DIM, col0, kc), acc);
  const int lane = threadIdx.x & 31;
  const int col = col0 + (lane & 15);
  const int h = lane >> 4;
  const float bv   = (col < NOUT) ? bias[col] : 0.0f;
  const float keep = (col < NOUT) ? 1.0f : 0.0f;
#pragma unroll
  for (int r = 0; r < 8; ++r) {
    const int row = row0 + r + 8 * h;
    hat[(size_t)row * NPAD + col] = keep * (acc[r] + bv);
  }
}

__global__ void k_zero(float* __restrict__ p, int n) {
  int i = blockIdx.x * blockDim.x + threadIdx.x;
  if (i < n) p[i] = 0.0f;
}

// Per (i,c): mean/center/norm of hat_m row; stage f16 copies:
//  mH [c][i][DPAD]   raw       (B^T layout for m_dot_v, k = d contiguous)
//  mcH[c][i][DPAD]   centered  (B^T layout for pearson)
//  mTH[c][d][I]      raw, transposed (B^T layout for hat_v, k = i contiguous)
__global__ void k_mstats(const float* __restrict__ hat, _Float16* __restrict__ mH,
                         _Float16* __restrict__ mcH, _Float16* __restrict__ mTH,
                         float* __restrict__ mnorm) {
  const int id = blockIdx.x;               // i*C + c
  const int c = id % NUM_CAPS, i = id / NUM_CAPS;
  const int lane = threadIdx.x;
  const float* src = hat + (size_t)i * NPAD + c * DIM_CAPS;
  float s = 0.f;
  for (int d = lane; d < DIM_CAPS; d += 32) s += src[d];
  const float mean = wave_sum(s) * (1.0f / (float)DIM_CAPS);
  _Float16* mh  = mH  + ((size_t)c * IN_CAPS + i) * DPAD;
  _Float16* mch = mcH + ((size_t)c * IN_CAPS + i) * DPAD;
  _Float16* mth = mTH + (size_t)c * DPAD * IN_CAPS + i;
  float ss = 0.f;
  for (int d = lane; d < DPAD; d += 32) {
    const float xv = (d < DIM_CAPS) ? src[d] : 0.f;
    const float xc = (d < DIM_CAPS) ? (xv - mean) : 0.f;
    ss += xc * xc;
    mh[d]  = (_Float16)xv;
    mch[d] = (_Float16)xc;
    mth[(size_t)d * IN_CAPS] = (_Float16)xv;
  }
  ss = wave_sum(ss);
  if (lane == 0) mnorm[c * IN_CAPS + i] = sqrtf(ss);
}

// tmp_q[q][c][0:153] = hat_q ; pads zero
__global__ void k_init_tmpq(const float* __restrict__ hat, float* __restrict__ tmpq) {
  int idx = blockIdx.x * blockDim.x + threadIdx.x;
  if (idx >= NQ * NUM_CAPS * DPAD) return;
  const int d = idx % DPAD, qc = idx / DPAD;
  const int c = qc % NUM_CAPS, q = qc / NUM_CAPS;
  tmpq[idx] = (d < DIM_CAPS)
            ? hat[(size_t)(IN_CAPS + q) * NPAD + c * DIM_CAPS + d] : 0.f;
}

// Per (q,c): center tmp_q row -> qcH[c][q][DPAD] f16, qnorm[c][q]
__global__ void k_qstats(const float* __restrict__ tmpq, _Float16* __restrict__ qcH,
                         float* __restrict__ qnorm) {
  const int id = blockIdx.x;               // q*C + c
  const int c = id % NUM_CAPS, q = id / NUM_CAPS;
  const int lane = threadIdx.x;
  const float* src = tmpq + (size_t)id * DPAD;
  float s = 0.f;
  for (int d = lane; d < DIM_CAPS; d += 32) s += src[d];
  const float mean = wave_sum(s) * (1.0f / (float)DIM_CAPS);
  _Float16* dst = qcH + ((size_t)c * NQ + q) * DPAD;
  float ss = 0.f;
  for (int d = lane; d < DPAD; d += 32) {
    const float xc = (d < DIM_CAPS) ? (src[d] - mean) : 0.f;
    ss += xc * xc;
    dst[d] = (_Float16)xc;
  }
  ss = wave_sum(ss);
  if (lane == 0) qnorm[c * NQ + q] = sqrtf(ss);
}

// p[q][i][c] = tanh( (qc . mc) / (||mc||*||qc|| + eps) )  via WMMA over D
__global__ void k_pearson(const _Float16* __restrict__ qcH, const _Float16* __restrict__ mcH,
                          const float* __restrict__ qnorm, const float* __restrict__ mnorm,
                          float* __restrict__ p) {
  const int it = blockIdx.x, qt = blockIdx.y, c = blockIdx.z;
  const _Float16* A = qcH + (size_t)c * NQ * DPAD;
  const _Float16* B = mcH + (size_t)c * IN_CAPS * DPAD;
  v8f acc = {};
#pragma unroll
  for (int kc = 0; kc < DPAD; kc += 32)
    acc = wmma_f16(fragA_f16(A, DPAD, qt * 16, kc),
                   fragB_f16(B, DPAD, it * 16, kc), acc);
  const int lane = threadIdx.x & 31;
  const int i = it * 16 + (lane & 15);
  const int h = lane >> 4;
  const float nm = mnorm[c * IN_CAPS + i];
#pragma unroll
  for (int r = 0; r < 8; ++r) {
    const int q = qt * 16 + r + 8 * h;
    const float den = nm * qnorm[c * NQ + q] + EPSF;
    p[((size_t)q * IN_CAPS + i) * NUM_CAPS + c] = tanhf(acc[r] / den);
  }
}

// dspH[c][q][i] = softmax_c(a[q][i][:]) + p[q][i][c]   (f16, A-layout for hat_v)
__global__ void k_softmax_dsp(const float* __restrict__ a, const float* __restrict__ p,
                              _Float16* __restrict__ dspH) {
  int idx = blockIdx.x * blockDim.x + threadIdx.x;   // q*I + i
  if (idx >= NQ * IN_CAPS) return;
  const float* av = a + (size_t)idx * NUM_CAPS;
  const float* pv = p + (size_t)idx * NUM_CAPS;
  float mx = av[0];
#pragma unroll
  for (int c = 1; c < NUM_CAPS; ++c) mx = fmaxf(mx, av[c]);
  float e[NUM_CAPS], s = 0.f;
#pragma unroll
  for (int c = 0; c < NUM_CAPS; ++c) { e[c] = __expf(av[c] - mx); s += e[c]; }
  const float inv = 1.0f / s;
#pragma unroll
  for (int c = 0; c < NUM_CAPS; ++c)
    dspH[(size_t)c * NQ * IN_CAPS + idx] = (_Float16)(e[c] * inv + pv[c]);
}

// hv[q][c][d] = sum_i dsp[q][i][c] * hat_m[i][c][d]   via WMMA over I
__global__ void k_hatv(const _Float16* __restrict__ dspH, const _Float16* __restrict__ mTH,
                       float* __restrict__ hv) {
  const int dt = blockIdx.x, qt = blockIdx.y, c = blockIdx.z;
  const _Float16* A = dspH + (size_t)c * NQ * IN_CAPS;
  const _Float16* B = mTH  + (size_t)c * DPAD * IN_CAPS;
  v8f acc = {};
#pragma unroll
  for (int kc = 0; kc < IN_CAPS; kc += 32)
    acc = wmma_f16(fragA_f16(A, IN_CAPS, qt * 16, kc),
                   fragB_f16(B, IN_CAPS, dt * 16, kc), acc);
  const int lane = threadIdx.x & 31;
  const int d = dt * 16 + (lane & 15);
  const int h = lane >> 4;
#pragma unroll
  for (int r = 0; r < 8; ++r) {
    const int q = qt * 16 + r + 8 * h;
    hv[((size_t)q * NUM_CAPS + c) * DPAD + d] = acc[r];
  }
}

// v = squash(hv); if final, write trimmed output [Q][765]
__global__ void k_squash(const float* __restrict__ hv, float* __restrict__ v,
                         _Float16* __restrict__ vH, float* __restrict__ out, int fin) {
  const int id = blockIdx.x;               // q*C + c
  const int c = id % NUM_CAPS, q = id / NUM_CAPS;
  const int lane = threadIdx.x;
  const float* src = hv + (size_t)id * DPAD;
  float ss = 0.f;
  for (int d = lane; d < DPAD; d += 32) { const float x = src[d]; ss += x * x; }
  ss = wave_sum(ss);
  const float scale = (ss / (1.0f + ss)) / sqrtf(ss + EPSF);
  if (fin) {
    for (int d = lane; d < DIM_CAPS; d += 32)
      out[(size_t)q * NOUT + c * DIM_CAPS + d] = scale * src[d];
  } else {
    float* vd = v + (size_t)id * DPAD;
    _Float16* vh = vH + ((size_t)c * NQ + q) * DPAD;
    for (int d = lane; d < DPAD; d += 32) {
      const float x = scale * src[d];
      vd[d] = x;
      vh[d] = (_Float16)x;
    }
  }
}

// a[q][i][c] += p[q][i][c] * (hat_m[i][c][:] . v[q][c][:])   via WMMA over D
__global__ void k_mdv(const _Float16* __restrict__ vH, const _Float16* __restrict__ mH,
                      const float* __restrict__ p, float* __restrict__ a) {
  const int it = blockIdx.x, qt = blockIdx.y, c = blockIdx.z;
  const _Float16* A = vH + (size_t)c * NQ * DPAD;
  const _Float16* B = mH + (size_t)c * IN_CAPS * DPAD;
  v8f acc = {};
#pragma unroll
  for (int kc = 0; kc < DPAD; kc += 32)
    acc = wmma_f16(fragA_f16(A, DPAD, qt * 16, kc),
                   fragB_f16(B, DPAD, it * 16, kc), acc);
  const int lane = threadIdx.x & 31;
  const int i = it * 16 + (lane & 15);
  const int h = lane >> 4;
#pragma unroll
  for (int r = 0; r < 8; ++r) {
    const int q = qt * 16 + r + 8 * h;
    const size_t off = ((size_t)q * IN_CAPS + i) * NUM_CAPS + c;
    a[off] += p[off] * acc[r];
  }
}

__global__ void k_tmpq_upd(float* __restrict__ tmpq, const float* __restrict__ v) {
  int idx = blockIdx.x * blockDim.x + threadIdx.x;
  if (idx < NQ * NUM_CAPS * DPAD) tmpq[idx] = (tmpq[idx] + v[idx]) * 0.5f;
}

// ---------------- host-side orchestration ----------------------------------
extern "C" void kernel_launch(void* const* d_in, const int* in_sizes, int n_in,
                              void* d_out, int out_size, void* d_ws, size_t ws_size,
                              hipStream_t stream) {
  const float* m    = (const float*)d_in[0];
  const float* q    = (const float*)d_in[1];
  const float* W    = (const float*)d_in[2];
  const float* bias = (const float*)d_in[3];
  float* out = (float*)d_out;

  char* ws = (char*)d_ws;
  size_t off = 0;
  auto carve = [&](size_t bytes) -> void* {
    void* p = ws + off;
    off = (off + bytes + 255) & ~(size_t)255;
    return p;
  };
  _Float16*  AH    = (_Float16*) carve((size_t)NROWS * IN_DIM * 2);
  _Float16*  WH    = (_Float16*) carve((size_t)NPAD * IN_DIM * 2);
  float*     hat   = (float*)    carve((size_t)NROWS * NPAD * 4);
  _Float16*  mH    = (_Float16*) carve((size_t)NUM_CAPS * IN_CAPS * DPAD * 2);
  _Float16*  mcH   = (_Float16*) carve((size_t)NUM_CAPS * IN_CAPS * DPAD * 2);
  _Float16*  mTH   = (_Float16*) carve((size_t)NUM_CAPS * DPAD * IN_CAPS * 2);
  float*     mnorm = (float*)    carve((size_t)NUM_CAPS * IN_CAPS * 4);
  float*     tmpq  = (float*)    carve((size_t)NQ * NUM_CAPS * DPAD * 4);
  _Float16*  qcH   = (_Float16*) carve((size_t)NUM_CAPS * NQ * DPAD * 2);
  float*     qnorm = (float*)    carve((size_t)NUM_CAPS * NQ * 4);
  float*     pbuf  = (float*)    carve((size_t)NQ * IN_CAPS * NUM_CAPS * 4);
  float*     abuf  = (float*)    carve((size_t)NQ * IN_CAPS * NUM_CAPS * 4);
  _Float16*  dspH  = (_Float16*) carve((size_t)NUM_CAPS * NQ * IN_CAPS * 2);
  float*     hv    = (float*)    carve((size_t)NQ * NUM_CAPS * DPAD * 4);
  float*     vbuf  = (float*)    carve((size_t)NQ * NUM_CAPS * DPAD * 4);
  _Float16*  vH    = (_Float16*) carve((size_t)NUM_CAPS * NQ * DPAD * 2);

  const int nA  = NQ * IN_CAPS * NUM_CAPS;       // 163840
  const int nTQ = NQ * NUM_CAPS * DPAD;          // 204800
  const int nAH = NROWS * IN_DIM;                // 294912
  const dim3 gW(IN_DIM / 16, NPAD / 16);         // 48 x 48 transpose tiles
  const dim3 gP(NPAD / 16, NROWS / 16);          // 48 x 24
  const dim3 gPear(IN_CAPS / 16, NQ / 16, NUM_CAPS);   // 8 x 16 x 5
  const dim3 gHV(DPAD / 16, NQ / 16, NUM_CAPS);        // 10 x 16 x 5

  // one-shot staging: f16 operands with zero-padded edges
  k_prepA<<<(nAH + 255) / 256, 256, 0, stream>>>(m, q, AH);
  k_prepW<<<gW, 256, 0, stream>>>(W, WH);
  k_zero<<<(nA + 255) / 256, 256, 0, stream>>>(abuf, nA);

  // projections + per-capsule staging
  k_proj<<<gP, 32, 0, stream>>>(AH, WH, bias, hat);
  k_mstats<<<IN_CAPS * NUM_CAPS, 32, 0, stream>>>(hat, mH, mcH, mTH, mnorm);
  k_init_tmpq<<<(nTQ + 255) / 256, 256, 0, stream>>>(hat, tmpq);

  // initial similarity
  k_qstats<<<NQ * NUM_CAPS, 32, 0, stream>>>(tmpq, qcH, qnorm);
  k_pearson<<<gPear, 32, 0, stream>>>(qcH, mcH, qnorm, mnorm, pbuf);

  // NUM_ROUTING - 1 = 2 refinement rounds
  for (int it = 0; it < 2; ++it) {
    k_softmax_dsp<<<(NQ * IN_CAPS + 255) / 256, 256, 0, stream>>>(abuf, pbuf, dspH);
    k_hatv<<<gHV, 32, 0, stream>>>(dspH, mTH, hv);
    k_squash<<<NQ * NUM_CAPS, 32, 0, stream>>>(hv, vbuf, vH, out, 0);
    k_mdv<<<gPear, 32, 0, stream>>>(vH, mH, pbuf, abuf);
    k_tmpq_upd<<<(nTQ + 255) / 256, 256, 0, stream>>>(tmpq, vbuf);
    k_qstats<<<NQ * NUM_CAPS, 32, 0, stream>>>(tmpq, qcH, qnorm);
    k_pearson<<<gPear, 32, 0, stream>>>(qcH, mcH, qnorm, mnorm, pbuf);
  }

  // final aggregation -> output [Q, 765]
  k_softmax_dsp<<<(NQ * IN_CAPS + 255) / 256, 256, 0, stream>>>(abuf, pbuf, dspH);
  k_hatv<<<gHV, 32, 0, stream>>>(dspH, mTH, hv);
  k_squash<<<NQ * NUM_CAPS, 32, 0, stream>>>(hv, vbuf, vH, out, 1);
}
